// Master_Slave_MOE_8143257993605
// MI455X (gfx1250) — matrix operand order repaired and checked by reading, not compiled, not run on hardware
//
#include <hip/hip_runtime.h>
#include <cstdint>

#define TOK 16384
#define DIM_ 512
#define HID_ 2048
#define NEXP 8

typedef __bf16 bf16;
typedef __attribute__((ext_vector_type(16))) __bf16 v16bf;
typedef __attribute__((ext_vector_type(8))) float v8f;

struct U32B { uint4 lo, hi; };

__device__ __forceinline__ v16bf frag_from_lds(const bf16* p0, const bf16* p1) {
  U32B u;
  u.lo = *reinterpret_cast<const uint4*>(p0);
  u.hi = *reinterpret_cast<const uint4*>(p1);
  return __builtin_bit_cast(v16bf, u);
}

// Split workgroup barrier WITHOUT the loadcnt flush that __syncthreads implies.
// Only LDS (dscnt) must be drained for cross-wave visibility; outstanding global
// prefetch loads keep flying and are waited on at their register use sites.
__device__ __forceinline__ void wg_sync_lds() {
  asm volatile(
      "s_wait_dscnt 0x0\n"
      "s_barrier_signal -1\n"
      "s_barrier_wait -1"
      ::: "memory");
}

// branch-free tanh-approx GELU:
// tanh(u) = sign(u) * (1 - z)/(1 + z),  z = exp(-2|u|) = exp2(-2*log2(e)*|u|)
__device__ __forceinline__ float gelu_fast(float x) {
  float u = 0.7978845608028654f * (x + 0.044715f * x * x * x);
  float a = __builtin_fabsf(u);
  float z = __builtin_amdgcn_exp2f(-2.885390081777927f * a);  // exp(-2a), ->0 for big a
  float t = (1.0f - z) * __builtin_amdgcn_rcpf(1.0f + z);
  t = __builtin_copysignf(t, u);
  return 0.5f * x * (1.0f + t);
}

// ---------------- elementwise cast f32 -> bf16 ----------------
__global__ void cast_f32_bf16(const float* __restrict__ in, bf16* __restrict__ out, int n) {
  int i = blockIdx.x * blockDim.x + threadIdx.x;
  int stride = gridDim.x * blockDim.x;
  for (; i < n; i += stride) out[i] = (bf16)in[i];
}

// ---------------- router: one wave per token ----------------
__global__ void router_kernel(const float* __restrict__ x, const float* __restrict__ gw,
                              int* __restrict__ topi, float* __restrict__ gates) {
  int wave = threadIdx.x >> 5;
  int lane = threadIdx.x & 31;
  int t = blockIdx.x * 8 + wave;
  if (t >= TOK) return;
  const float* xr = x + (size_t)t * DIM_;
  float acc[NEXP];
#pragma unroll
  for (int e = 0; e < NEXP; ++e) acc[e] = 0.f;
  for (int i = 0; i < DIM_ / 32; ++i) {
    int d = i * 32 + lane;
    float xv = xr[d];
    const float* g = gw + (size_t)d * NEXP;
#pragma unroll
    for (int e = 0; e < NEXP; ++e) acc[e] += xv * g[e];
  }
#pragma unroll
  for (int e = 0; e < NEXP; ++e) {
#pragma unroll
    for (int s = 16; s > 0; s >>= 1) acc[e] += __shfl_xor(acc[e], s, 32);
  }
  // top-2 (first occurrence wins ties, like lax.top_k)
  int i0 = 0; float v0 = acc[0];
#pragma unroll
  for (int e = 1; e < NEXP; ++e) if (acc[e] > v0) { v0 = acc[e]; i0 = e; }
  int i1 = -1; float v1 = -3.4e38f;
#pragma unroll
  for (int e = 0; e < NEXP; ++e) if (e != i0 && acc[e] > v1) { v1 = acc[e]; i1 = e; }
  if (lane == 0) {
    float ex = __expf(v1 - v0);
    float denom = 1.0f + ex;
    topi[t * 2] = i0; topi[t * 2 + 1] = i1;
    gates[t * 2] = 1.0f / denom; gates[t * 2 + 1] = ex / denom;
  }
}

// ---------------- deterministic counting sort ----------------
__global__ void count_kernel(const int* __restrict__ topi, int* __restrict__ blockCounts) {
  __shared__ int cnt[NEXP];
  if (threadIdx.x < NEXP) cnt[threadIdx.x] = 0;
  __syncthreads();
  int t = blockIdx.x * 256 + threadIdx.x;
  atomicAdd(&cnt[topi[t * 2]], 1);
  atomicAdd(&cnt[topi[t * 2 + 1]], 1);
  __syncthreads();
  if (threadIdx.x < NEXP) blockCounts[blockIdx.x * NEXP + threadIdx.x] = cnt[threadIdx.x];
}

__global__ void scan_kernel(const int* __restrict__ blockCounts, int* __restrict__ blockBase,
                            int* __restrict__ expertOff, int* __restrict__ expertCount) {
  __shared__ int ecnt[NEXP];
  int e = threadIdx.x;
  if (e < NEXP) {
    int run = 0;
    for (int b = 0; b < TOK / 256; ++b) { blockBase[b * NEXP + e] = run; run += blockCounts[b * NEXP + e]; }
    ecnt[e] = run;
    expertCount[e] = run;
  }
  __syncthreads();
  if (e == 0) {
    int off = 0;
    for (int k = 0; k < NEXP; ++k) { expertOff[k] = off; off += ecnt[k]; }
  }
  __syncthreads();
  if (e < NEXP) {
    int off = expertOff[e];
    for (int b = 0; b < TOK / 256; ++b) blockBase[b * NEXP + e] += off;
  }
}

__global__ void assign_kernel(const int* __restrict__ topi, const int* __restrict__ blockBase,
                              int* __restrict__ assignTok, int* __restrict__ tokenSlot) {
  if (threadIdx.x != 0) return;
  int ptr[NEXP];
  for (int e = 0; e < NEXP; ++e) ptr[e] = blockBase[blockIdx.x * NEXP + e];
  int t0 = blockIdx.x * 256;
  for (int i = 0; i < 256; ++i) {
    int t = t0 + i;
    for (int j = 0; j < 2; ++j) {
      int e = topi[t * 2 + j];
      int s = ptr[e]++;
      assignTok[s] = t;
      tokenSlot[t * 2 + j] = s;
    }
  }
}

// ---------------- WMMA bf16 GEMM, 128x128x32 tile, 8 waves ----------------
// FC1: out = gelu(A @ W + b) stored bf16   (A lda = K = DIM,  N = HID)
// FC2: out = A @ W + b stored f32          (A lda = K = HID,  N = DIM)
// EXPERT: rows/offset from routing tables; FC1 gathers x rows via assignTok,
// FC2 reads/writes slot-indexed buffers directly.
template <bool EXPERT, bool FC1>
__global__ __launch_bounds__(256) void moe_gemm(
    const bf16* __restrict__ A, const bf16* __restrict__ Wb, const float* __restrict__ bias,
    void* __restrict__ outp, const int* __restrict__ assignTok,
    const int* __restrict__ expertOff, const int* __restrict__ expertCount) {
  constexpr int K = FC1 ? DIM_ : HID_;
  constexpr int N = FC1 ? HID_ : DIM_;
  const int e = EXPERT ? blockIdx.z : 0;
  const int rows = EXPERT ? expertCount[e] : TOK;
  const int rowOff = EXPERT ? expertOff[e] : 0;
  const int tileM = blockIdx.x * 128;
  if (tileM >= rows) return;
  const int colBase = blockIdx.y * 128;
  const bf16* W = Wb + (size_t)e * DIM_ * HID_;
  const float* bs = bias + (size_t)e * N;

  __shared__ bf16 Al[128 * 40];   // A tile, row-major, padded stride 40
  __shared__ bf16 Bt[128 * 40];   // W tile transposed: [n][k], padded stride 40

  const int tid = threadIdx.x;
  const int lane = tid & 31;
  const int wv = tid >> 5;
  const int wm = wv & 3;       // wave M position (0..3) -> 32 rows each
  const int wn = wv >> 2;      // wave N position (0..1) -> 64 cols each
  const int lanelow = lane & 15;
  const int khalf = lane >> 4;

  // A staging: this thread fills rows m0 and m0+64, 8 bf16 at column kc
  const int m0 = tid >> 2;
  const int kc = (tid & 3) * 8;
  const int m1 = m0 + 64;
  int ar0 = tileM + m0, ar1 = tileM + m1;
  int cr0 = ar0 < rows ? ar0 : rows - 1;
  int cr1 = ar1 < rows ? ar1 : rows - 1;
  size_t asrc0, asrc1;
  if (EXPERT && FC1) {            // gather x rows via token list
    asrc0 = (size_t)assignTok[rowOff + cr0];
    asrc1 = (size_t)assignTok[rowOff + cr1];
  } else if (EXPERT) {            // slot-indexed h buffer
    asrc0 = (size_t)(rowOff + cr0);
    asrc1 = (size_t)(rowOff + cr1);
  } else {
    asrc0 = (size_t)cr0;
    asrc1 = (size_t)cr1;
  }
  const uint4* ag0 = reinterpret_cast<const uint4*>(A + asrc0 * K + kc);
  const uint4* ag1 = reinterpret_cast<const uint4*>(A + asrc1 * K + kc);

  // B staging: thread handles (k = tid>>4, n = (tid&15)*8) and k+16
  const int bk0 = tid >> 4;
  const int bn0 = (tid & 15) * 8;

  v8f acc[2][4];
#pragma unroll
  for (int ni = 0; ni < 4; ++ni) {
    float bv = bs[colBase + wn * 64 + ni * 16 + lanelow];
#pragma unroll
    for (int mi = 0; mi < 2; ++mi)
#pragma unroll
      for (int r = 0; r < 8; ++r) acc[mi][ni][r] = bv;
  }

  // register double-buffer: prime first tile
  uint4 a0 = ag0[0];
  uint4 a1 = ag1[0];
  uint4 w0 = *reinterpret_cast<const uint4*>(W + (size_t)bk0 * N + colBase + bn0);
  uint4 w1 = *reinterpret_cast<const uint4*>(W + (size_t)(bk0 + 16) * N + colBase + bn0);

  for (int kk = 0; kk < K; kk += 32) {
    wg_sync_lds();
    // stage current tile to LDS (from registers)
    *reinterpret_cast<uint4*>(&Al[m0 * 40 + kc]) = a0;
    *reinterpret_cast<uint4*>(&Al[m1 * 40 + kc]) = a1;
    {
      const bf16* w0e = reinterpret_cast<const bf16*>(&w0);
      const bf16* w1e = reinterpret_cast<const bf16*>(&w1);
#pragma unroll
      for (int i = 0; i < 8; ++i) {
        Bt[(bn0 + i) * 40 + bk0] = w0e[i];
        Bt[(bn0 + i) * 40 + bk0 + 16] = w1e[i];
      }
    }
    // issue next tile's global loads now; their wait lands at next iteration's
    // LDS stores (register dependence), overlapping this iteration's WMMAs
    {
      int kn = (kk + 32 < K) ? (kk + 32) : 0;   // clamped dummy on last iter
      a0 = ag0[kn >> 3];
      a1 = ag1[kn >> 3];
      w0 = *reinterpret_cast<const uint4*>(W + (size_t)(kn + bk0) * N + colBase + bn0);
      w1 = *reinterpret_cast<const uint4*>(W + (size_t)(kn + bk0 + 16) * N + colBase + bn0);
    }
    wg_sync_lds();

    // fragments per ISA layout:
    // A 16x32: lane m=lanelow, elems 0..7 = K[khalf*8..], elems 8..15 = K[16+khalf*8..]
    v16bf af[2];
#pragma unroll
    for (int mi = 0; mi < 2; ++mi) {
      int m = wm * 32 + mi * 16 + lanelow;
      af[mi] = frag_from_lds(&Al[m * 40 + khalf * 8], &Al[m * 40 + 16 + khalf * 8]);
    }
    // B 32x16: lane n=lanelow, elems 0..7 = K[khalf*8..], elems 8..15 = K[16+khalf*8..]
    v16bf bfr[4];
#pragma unroll
    for (int ni = 0; ni < 4; ++ni) {
      int n = wn * 64 + ni * 16 + lanelow;
      bfr[ni] = frag_from_lds(&Bt[n * 40 + khalf * 8], &Bt[n * 40 + 16 + khalf * 8]);
    }
#pragma unroll
    for (int mi = 0; mi < 2; ++mi)
#pragma unroll
      for (int ni = 0; ni < 4; ++ni)
        acc[mi][ni] = __builtin_amdgcn_wmma_f32_16x16x32_bf16(
            false, af[mi], false, bfr[ni], (short)0, acc[mi][ni], false, false);
  }

  // epilogue; C/D layout: elem r, lanes0-15 -> M=r, lanes16-31 -> M=r+8, N=lanelow
#pragma unroll
  for (int mi = 0; mi < 2; ++mi) {
#pragma unroll
    for (int r = 0; r < 8; ++r) {
      int grow = tileM + wm * 32 + mi * 16 + r + 8 * khalf;
      if (grow >= rows) continue;
      size_t orow = (size_t)(rowOff + grow);
#pragma unroll
      for (int ni = 0; ni < 4; ++ni) {
        int gcol = colBase + wn * 64 + ni * 16 + lanelow;
        float v = acc[mi][ni][r];
        if (FC1) {
          ((bf16*)outp)[orow * HID_ + gcol] = (bf16)gelu_fast(v);
        } else {
          ((float*)outp)[orow * DIM_ + gcol] = v;
        }
      }
    }
  }
}

// ---------------- final combine: out = master + g0*eo[s0] + g1*eo[s1] ----------------
__global__ void combine_kernel(float* __restrict__ out, const float* __restrict__ eo,
                               const int* __restrict__ tokenSlot, const float* __restrict__ gates) {
  int idx = blockIdx.x * blockDim.x + threadIdx.x;  // over TOK * DIM/4
  if (idx >= TOK * (DIM_ / 4)) return;
  int t = idx >> 7;        // DIM/4 = 128 float4 per token
  int c = idx & 127;
  int s0 = tokenSlot[t * 2], s1 = tokenSlot[t * 2 + 1];
  float g0 = gates[t * 2], g1 = gates[t * 2 + 1];
  float4 a = reinterpret_cast<const float4*>(eo)[(size_t)s0 * 128 + c];
  float4 b = reinterpret_cast<const float4*>(eo)[(size_t)s1 * 128 + c];
  float4* o = reinterpret_cast<float4*>(out) + idx;
  float4 m = *o;
  m.x += g0 * a.x + g1 * b.x;
  m.y += g0 * a.y + g1 * b.y;
  m.z += g0 * a.z + g1 * b.z;
  m.w += g0 * a.w + g1 * b.w;
  *o = m;
}

extern "C" void kernel_launch(void* const* d_in, const int* in_sizes, int n_in,
                              void* d_out, int out_size, void* d_ws, size_t ws_size,
                              hipStream_t stream) {
  (void)in_sizes; (void)n_in; (void)out_size; (void)ws_size;
  const float* x   = (const float*)d_in[0];
  const float* gw  = (const float*)d_in[1];
  const float* mw1 = (const float*)d_in[2];
  const float* mb1 = (const float*)d_in[3];
  const float* mw2 = (const float*)d_in[4];
  const float* mb2 = (const float*)d_in[5];
  const float* ew1 = (const float*)d_in[6];
  const float* eb1 = (const float*)d_in[7];
  const float* ew2 = (const float*)d_in[8];
  const float* eb2 = (const float*)d_in[9];
  float* out = (float*)d_out;

  char* ws = (char*)d_ws;
  size_t off = 0;
  auto alloc = [&](size_t bytes) { char* p = ws + off; off = (off + bytes + 255) & ~(size_t)255; return p; };
  bf16* x_bf      = (bf16*)alloc((size_t)TOK * DIM_ * 2);
  bf16* mw1_bf    = (bf16*)alloc((size_t)DIM_ * HID_ * 2);
  bf16* mw2_bf    = (bf16*)alloc((size_t)DIM_ * HID_ * 2);
  bf16* ew1_bf    = (bf16*)alloc((size_t)NEXP * DIM_ * HID_ * 2);
  bf16* ew2_bf    = (bf16*)alloc((size_t)NEXP * DIM_ * HID_ * 2);
  bf16* h_m       = (bf16*)alloc((size_t)TOK * HID_ * 2);
  bf16* h_e       = (bf16*)alloc((size_t)2 * TOK * HID_ * 2);
  float* eo       = (float*)alloc((size_t)2 * TOK * DIM_ * 4);
  int* topi       = (int*)alloc((size_t)TOK * 2 * 4);
  float* gates    = (float*)alloc((size_t)TOK * 2 * 4);
  int* assignTok  = (int*)alloc((size_t)2 * TOK * 4);
  int* tokenSlot  = (int*)alloc((size_t)2 * TOK * 4);
  int* blockCounts = (int*)alloc((TOK / 256) * NEXP * 4);
  int* blockBase   = (int*)alloc((TOK / 256) * NEXP * 4);
  int* expertOff   = (int*)alloc(NEXP * 4);
  int* expertCount = (int*)alloc(NEXP * 4);

  // 1) casts to bf16
  cast_f32_bf16<<<2048, 256, 0, stream>>>(x, x_bf, TOK * DIM_);
  cast_f32_bf16<<<1024, 256, 0, stream>>>(mw1, mw1_bf, DIM_ * HID_);
  cast_f32_bf16<<<1024, 256, 0, stream>>>(mw2, mw2_bf, DIM_ * HID_);
  cast_f32_bf16<<<2048, 256, 0, stream>>>(ew1, ew1_bf, NEXP * DIM_ * HID_);
  cast_f32_bf16<<<2048, 256, 0, stream>>>(ew2, ew2_bf, NEXP * DIM_ * HID_);

  // 2) routing (deterministic)
  router_kernel<<<TOK / 8, 256, 0, stream>>>(x, gw, topi, gates);
  count_kernel<<<TOK / 256, 256, 0, stream>>>(topi, blockCounts);
  scan_kernel<<<1, 64, 0, stream>>>(blockCounts, blockBase, expertOff, expertCount);
  assign_kernel<<<TOK / 256, 64, 0, stream>>>(topi, blockBase, assignTok, tokenSlot);

  dim3 blk(256);
  // 3) master MLP
  moe_gemm<false, true><<<dim3(TOK / 128, HID_ / 128, 1), blk, 0, stream>>>(
      x_bf, mw1_bf, mb1, h_m, nullptr, nullptr, nullptr);
  moe_gemm<false, false><<<dim3(TOK / 128, DIM_ / 128, 1), blk, 0, stream>>>(
      h_m, mw2_bf, mb2, out, nullptr, nullptr, nullptr);
  // 4) routed experts (blocks past each expert's token count exit immediately)
  moe_gemm<true, true><<<dim3(TOK / 128, HID_ / 128, NEXP), blk, 0, stream>>>(
      x_bf, ew1_bf, eb1, h_e, assignTok, expertOff, expertCount);
  moe_gemm<true, false><<<dim3(TOK / 128, DIM_ / 128, NEXP), blk, 0, stream>>>(
      h_e, ew2_bf, eb2, eo, assignTok, expertOff, expertCount);

  // 5) gated combine
  combine_kernel<<<(TOK * (DIM_ / 4) + 255) / 256, 256, 0, stream>>>(out, eo, tokenSlot, gates);
}
